// SequenceParallelWanSelfAttention_39462159516023
// MI455X (gfx1250) — matrix operand order repaired and checked by reading, not compiled, not run on hardware
//
#include <hip/hip_runtime.h>

typedef __attribute__((ext_vector_type(16))) _Float16 v16h;
typedef __attribute__((ext_vector_type(8)))  _Float16 v8h;
typedef __attribute__((ext_vector_type(8)))  float    v8f;
typedef unsigned int u32x4 __attribute__((ext_vector_type(4)));
typedef int          i32x8 __attribute__((ext_vector_type(8)));
typedef int          i32x4 __attribute__((ext_vector_type(4)));

constexpr int L_  = 4096;
constexpr int D_  = 1536;
constexpr int NH  = 12;
constexpr int HD  = 128;

// Assemble a 16-half WMMA fragment from two contiguous 8-half chunks.
static __device__ __forceinline__ v16h ldAB(const _Float16* p0, const _Float16* p1) {
    union { v16h v; v8h h[2]; } u;
    u.h[0] = *(const v8h*)p0;
    u.h[1] = *(const v8h*)p1;
    return u.v;
}

// ---------------------------------------------------------------------------
// Tensor Data Mover: 2D f16 tile (tile1 rows x tile0 elems, row stride
// stride0 elems) from global -> LDS, optional LDS row padding.
// D# layout per CDNA5 ISA.
// ---------------------------------------------------------------------------
static __device__ __forceinline__ void tdm_load_2d_f16(
    unsigned int lds_addr, const void* gptr,
    unsigned int dim0, unsigned int tile0, unsigned int tile1,
    unsigned int stride0, int pad_en,
    unsigned int pad_interval_code, unsigned int pad_amount_code)
{
    unsigned long long ga = (unsigned long long)(size_t)gptr;
    union { u32x4 v; unsigned long long q[2]; } g0;
    g0.q[0] = 1ull /*count=1*/ | ((unsigned long long)lds_addr << 32);
    g0.q[1] = (ga & 0x01FFFFFFFFFFFFFFull) | (2ull << 62);   // type=2 ("image")

    union { i32x8 v; unsigned long long q[4]; } g1;
    unsigned long long flags = (1ull << 16);                 // data_size=1 -> 2 bytes
    if (pad_en)
        flags |= (1ull << 20) | ((unsigned long long)pad_interval_code << 22)
               | ((unsigned long long)pad_amount_code << 25);
    g1.q[0] = flags | (((unsigned long long)dim0 & 0xFFFFull) << 48);          // dim0[15:0] @ 63:48
    g1.q[1] = ((unsigned long long)dim0 >> 16)                                  // dim0[31:16] @ 79:64
            | ((unsigned long long)(1u << 20) << 16)                            // tensor_dim1 (large) @ 111:80
            | ((unsigned long long)tile0 << 48);                                // tile_dim0 @ 127:112
    g1.q[2] = (unsigned long long)tile1                                         // tile_dim1 @ 143:128
            | ((unsigned long long)stride0 << 32);                              // stride0[31:0] @ 191:160
    g1.q[3] = ((unsigned long long)stride0 >> 32) & 0xFFFFull;                  // stride0[47:32] @ 207:192

    const i32x4 z4 = {0, 0, 0, 0};
    const i32x8 z8 = {0, 0, 0, 0, 0, 0, 0, 0};
    __builtin_amdgcn_tensor_load_to_lds(g0.v, g1.v, z4, z4, z8, 0);
}

// ---------------------------------------------------------------------------
// GEMM: C[M,N] = A[M,K](f32) * W[N,K]^T(f32) + bias[N], f32 out.
// Block tile 128x128, BK=32, 8 waves, wave tile 32x64. f16 staging in LDS.
// ---------------------------------------------------------------------------
__global__ __launch_bounds__(256) void gemm_f16wmma_kernel(
    const float* __restrict__ A, const float* __restrict__ W,
    const float* __restrict__ bias, float* __restrict__ C,
    int M, int Nn, int K)
{
    constexpr int LDT = 56;                 // padded LDS stride (halves): 112B, 16B-aligned
    __shared__ _Float16 sA[128 * LDT];
    __shared__ _Float16 sB[128 * LDT];

    const int tid  = threadIdx.x;
    const int wid  = tid >> 5;
    const int lane = tid & 31;
    const int l16  = lane & 15;
    const int hi   = lane >> 4;
    const int wm   = wid & 3;               // 4 m-blocks of 32
    const int wn   = wid >> 2;              // 2 n-blocks of 64
    const int mBase = blockIdx.y * 128;
    const int nBase = blockIdx.x * 128;

    v8f acc[2][4];
    const v8f zf = {0.f, 0.f, 0.f, 0.f, 0.f, 0.f, 0.f, 0.f};
#pragma unroll
    for (int mt = 0; mt < 2; ++mt)
#pragma unroll
        for (int nt = 0; nt < 4; ++nt) acc[mt][nt] = zf;

    const int lr  = tid >> 1;               // 0..127 : tile row this thread loads
    const int lk0 = (tid & 1) * 16;         // 0 or 16 : k segment

    for (int kt = 0; kt < K; kt += 32) {
        __syncthreads();
        if (kt + 32 < K) {                  // L2 prefetch of next k-tile (global_prefetch_b8)
            __builtin_prefetch(A + (size_t)(mBase + lr) * K + kt + 32 + lk0, 0, 3);
            __builtin_prefetch(W + (size_t)(nBase + lr) * K + kt + 32 + lk0, 0, 3);
        }
        {   // stage A (convert f32 -> f16)
            const float* src = A + (size_t)(mBase + lr) * K + kt + lk0;
            v8h h0, h1;
#pragma unroll
            for (int e = 0; e < 8; ++e) { h0[e] = (_Float16)src[e]; h1[e] = (_Float16)src[8 + e]; }
            _Float16* dst = sA + lr * LDT + lk0;
            *(v8h*)dst = h0; *(v8h*)(dst + 8) = h1;
        }
        {   // stage B = W rows (column of B is W row -> direct copy)
            const float* src = W + (size_t)(nBase + lr) * K + kt + lk0;
            v8h h0, h1;
#pragma unroll
            for (int e = 0; e < 8; ++e) { h0[e] = (_Float16)src[e]; h1[e] = (_Float16)src[8 + e]; }
            _Float16* dst = sB + lr * LDT + lk0;
            *(v8h*)dst = h0; *(v8h*)(dst + 8) = h1;
        }
        __syncthreads();

        v16h af[2], bf[4];
#pragma unroll
        for (int mt = 0; mt < 2; ++mt) {
            const _Float16* ar = sA + (wm * 32 + mt * 16 + l16) * LDT;
            af[mt] = ldAB(ar + 8 * hi, ar + 16 + 8 * hi);
        }
#pragma unroll
        for (int nt = 0; nt < 4; ++nt) {
            const _Float16* br = sB + (wn * 64 + nt * 16 + l16) * LDT;
            bf[nt] = ldAB(br + 8 * hi, br + 16 + 8 * hi);
        }
#pragma unroll
        for (int mt = 0; mt < 2; ++mt)
#pragma unroll
            for (int nt = 0; nt < 4; ++nt)
                acc[mt][nt] = __builtin_amdgcn_wmma_f32_16x16x32_f16(
                    false, af[mt], false, bf[nt], (short)0, acc[mt][nt], false, false);
    }

#pragma unroll
    for (int mt = 0; mt < 2; ++mt) {
#pragma unroll
        for (int nt = 0; nt < 4; ++nt) {
            const int col = nBase + wn * 64 + nt * 16 + l16;
            const float b = bias[col];
#pragma unroll
            for (int rr = 0; rr < 8; ++rr) {
                const int row = mBase + wm * 32 + mt * 16 + rr + 8 * hi;
                C[(size_t)row * Nn + col] = acc[mt][nt][rr] + b;
            }
        }
    }
}

// ---------------------------------------------------------------------------
// RMSNorm (over D=1536) + 3D RoPE (c1=22,c2=21,c2=21), emit f16 head-major
// out[(head*L + t)*128 + j].  doNormRope==0 -> plain convert (for V).
// ---------------------------------------------------------------------------
__global__ __launch_bounds__(256) void rmsrope_kernel(
    const float* __restrict__ raw, const float* __restrict__ g,
    const float* __restrict__ fcos, const float* __restrict__ fsin,
    const int* __restrict__ grid_sizes, _Float16* __restrict__ outh,
    int doNormRope)
{
    const int t   = blockIdx.x;
    const int tid = threadIdx.x;
    __shared__ float red[8];
    __shared__ float stot;

    float rms = 1.0f;
    if (doNormRope) {
        float s = 0.f;
        for (int i = tid; i < D_; i += 256) { float v = raw[(size_t)t * D_ + i]; s += v * v; }
#pragma unroll
        for (int off = 16; off > 0; off >>= 1) s += __shfl_down(s, off, 32);
        if ((tid & 31) == 0) red[tid >> 5] = s;
        __syncthreads();
        if (tid == 0) { float tt = 0.f; for (int i = 0; i < 8; ++i) tt += red[i]; stot = tt; }
        __syncthreads();
        rms = rsqrtf(stot * (1.0f / (float)D_) + 1e-6f);
    }

    const int gf = grid_sizes[0], gh = grid_sizes[1], gw = grid_sizes[2];
    const int sl = gf * gh * gw;
    const bool doRope = doNormRope && (t < sl);
    int tf = 0, th = 0, tw = 0;
    if (doRope) { tf = t / (gh * gw); th = (t / gw) % gh; tw = t % gw; }

    for (int p = tid; p < D_ / 2; p += 256) {
        const int head = p >> 6;      // 64 pairs per head
        const int j    = p & 63;
        const int base = head * HD + 2 * j;
        float x0 = raw[(size_t)t * D_ + base];
        float x1 = raw[(size_t)t * D_ + base + 1];
        if (doNormRope) { x0 = x0 * rms * g[base]; x1 = x1 * rms * g[base + 1]; }
        float o0 = x0, o1 = x1;
        if (doRope) {
            const int coord = (j < 22) ? tf : ((j < 43) ? th : tw);
            const float fc = fcos[coord * 64 + j];
            const float fs = fsin[coord * 64 + j];
            o0 = x0 * fc - x1 * fs;
            o1 = x0 * fs + x1 * fc;
        }
        const size_t o = ((size_t)head * L_ + t) * HD + 2 * j;
        outh[o]     = (_Float16)o0;
        outh[o + 1] = (_Float16)o1;
    }
}

// ---------------------------------------------------------------------------
// Flash attention, one (head, 128-q-row) tile per workgroup, 8 waves,
// each wave owns 16 q rows. KT=64 key tile, online softmax, f32 acc.
// K tile staged via Tensor Data Mover (with LDS row padding); V tile
// transposed manually in parallel with the DMA.
// ---------------------------------------------------------------------------
__global__ __launch_bounds__(256) void flash_kernel(
    const _Float16* __restrict__ qh, const _Float16* __restrict__ kh,
    const _Float16* __restrict__ vh, const int* __restrict__ seq_lens,
    float* __restrict__ attn)
{
    constexpr int LDK = 136;   // 64x128 K tile, padded stride (halves) = 256B + 16B TDM pad
    constexpr int LDV = 72;    // 128x64 V^T tile + per-wave P scratch stride
    __shared__ _Float16 sK [64 * LDK];
    __shared__ _Float16 sVt[128 * LDV];
    __shared__ _Float16 sP [8 * 16 * LDV];

    const int head  = blockIdx.y;
    const int qBase = blockIdx.x * 128;
    const int tid   = threadIdx.x;
    const int wid   = tid >> 5;
    const int lane  = tid & 31;
    const int l16   = lane & 15;
    const int hi    = lane >> 4;
    const int seqLen = seq_lens[0];
    const float scale = 0.08838834764831845f;   // 1/sqrt(128)
    const v8f zf = {0.f, 0.f, 0.f, 0.f, 0.f, 0.f, 0.f, 0.f};
    const unsigned int sK_lds = (unsigned int)(size_t)(void*)&sK[0];

    // Q fragments: 16 rows x 128 (d), 4 k-chunks of 32 — loaded once from global
    const _Float16* qp = qh + ((size_t)head * L_ + qBase + wid * 16 + l16) * HD;
    v16h qa[4];
#pragma unroll
    for (int kc = 0; kc < 4; ++kc)
        qa[kc] = ldAB(qp + kc * 32 + 8 * hi, qp + kc * 32 + 16 + 8 * hi);

    v8f acc[8];
#pragma unroll
    for (int nt = 0; nt < 8; ++nt) acc[nt] = zf;
    float ms[8], ls[8];
#pragma unroll
    for (int r = 0; r < 8; ++r) { ms[r] = -3.0e38f; ls[r] = 0.f; }

    const int ldr  = tid >> 2;        // 0..63 : key row this thread loads
    const int lseg = tid & 3;         // 32-half segment of d

    for (int kb = 0; kb < L_ / 64; ++kb) {
        __syncthreads();              // previous tiles fully consumed
        if (wid == 0) {
            // TDM: 64 rows x 128 f16, row stride 128, +16B LDS pad per row
            // (pad_interval code 5 = 64 dwords, pad_amount code 3 = 4 dwords)
            tdm_load_2d_f16(sK_lds, kh + ((size_t)head * L_ + kb * 64) * HD,
                            /*dim0=*/HD, /*tile0=*/HD, /*tile1=*/64,
                            /*stride0=*/HD, /*pad_en=*/1, 5u, 3u);
        }
        {   // V tile: transpose into [d][key] while the K DMA is in flight
            const _Float16* src = vh + ((size_t)head * L_ + kb * 64 + ldr) * HD + lseg * 32;
#pragma unroll
            for (int q8 = 0; q8 < 4; ++q8) {
                v8h vv = *(const v8h*)(src + q8 * 8);
#pragma unroll
                for (int e = 0; e < 8; ++e)
                    sVt[(lseg * 32 + q8 * 8 + e) * LDV + ldr] = vv[e];
            }
        }
        if (wid == 0) __builtin_amdgcn_s_wait_tensorcnt(0);   // K tile landed
        __syncthreads();

        // S = Q K^T : 16 x 64, four 16x16 tiles, each over 4 k-chunks
        v8f st[4];
#pragma unroll
        for (int t4 = 0; t4 < 4; ++t4) st[t4] = zf;
#pragma unroll
        for (int t4 = 0; t4 < 4; ++t4) {
            const _Float16* kr = sK + (t4 * 16 + l16) * LDK;
#pragma unroll
            for (int kc = 0; kc < 4; ++kc) {
                v16h bf = ldAB(kr + kc * 32 + 8 * hi, kr + kc * 32 + 16 + 8 * hi);
                st[t4] = __builtin_amdgcn_wmma_f32_16x16x32_f16(
                    false, qa[kc], false, bf, (short)0, st[t4], false, false);
            }
        }

        // online softmax per owned row r (row m = r + 8*hi in the C layout)
        const int colBase = kb * 64;
#pragma unroll
        for (int r = 0; r < 8; ++r) {
            float rmax = -3.0e38f;
#pragma unroll
            for (int t4 = 0; t4 < 4; ++t4) {
                float sv = st[t4][r] * scale;
                if (colBase + t4 * 16 + l16 >= seqLen) sv = -3.0e38f;
                st[t4][r] = sv;
                rmax = fmaxf(rmax, sv);
            }
#pragma unroll
            for (int m2 = 1; m2 <= 8; m2 <<= 1)   // stays within 16-lane half
                rmax = fmaxf(rmax, __shfl_xor(rmax, m2, 32));
            const float mnew  = fmaxf(ms[r], rmax);
            const float alpha = __expf(ms[r] - mnew);
            ms[r] = mnew;
            float psum = 0.f;
#pragma unroll
            for (int t4 = 0; t4 < 4; ++t4) {
                const float pv = __expf(st[t4][r] - mnew);
                st[t4][r] = pv;
                psum += pv;
            }
#pragma unroll
            for (int m2 = 1; m2 <= 8; m2 <<= 1)
                psum += __shfl_xor(psum, m2, 32);
            ls[r] = ls[r] * alpha + psum;
#pragma unroll
            for (int nt = 0; nt < 8; ++nt) acc[nt][r] *= alpha;
        }

        // C-layout P -> A-layout via per-wave LDS bounce
        _Float16* pbuf = sP + wid * 16 * LDV;
#pragma unroll
        for (int r = 0; r < 8; ++r) {
            const int m = r + 8 * hi;
#pragma unroll
            for (int t4 = 0; t4 < 4; ++t4)
                pbuf[m * LDV + t4 * 16 + l16] = (_Float16)st[t4][r];
        }

        // acc += P * V   (2 k-steps of 32 keys, 8 d-tiles)
        const _Float16* prow = pbuf + l16 * LDV;
#pragma unroll
        for (int ks = 0; ks < 2; ++ks) {
            v16h pf = ldAB(prow + ks * 32 + 8 * hi, prow + ks * 32 + 16 + 8 * hi);
#pragma unroll
            for (int nt = 0; nt < 8; ++nt) {
                const _Float16* vr = sVt + (nt * 16 + l16) * LDV + ks * 32 + 8 * hi;
                v16h bf = ldAB(vr, vr + 16);
                acc[nt] = __builtin_amdgcn_wmma_f32_16x16x32_f16(
                    false, pf, false, bf, (short)0, acc[nt], false, false);
            }
        }
    }

    // epilogue: normalize and scatter into [L, D] f32 (input to O-projection)
#pragma unroll
    for (int r = 0; r < 8; ++r) {
        const int qRow = qBase + wid * 16 + r + 8 * hi;
        const float inv = 1.0f / ls[r];
#pragma unroll
        for (int nt = 0; nt < 8; ++nt)
            attn[(size_t)qRow * D_ + head * HD + nt * 16 + l16] = acc[nt][r] * inv;
    }
}

// ---------------------------------------------------------------------------
extern "C" void kernel_launch(void* const* d_in, const int* in_sizes, int n_in,
                              void* d_out, int out_size, void* d_ws, size_t ws_size,
                              hipStream_t stream) {
    const float* x    = (const float*)d_in[0];
    const int*   seql = (const int*)  d_in[1];
    const int*   gs   = (const int*)  d_in[2];
    const float* fcos = (const float*)d_in[3];
    const float* fsin = (const float*)d_in[4];
    const float* Wq = (const float*)d_in[5];  const float* bq = (const float*)d_in[6];
    const float* Wk = (const float*)d_in[7];  const float* bk = (const float*)d_in[8];
    const float* Wv = (const float*)d_in[9];  const float* bv = (const float*)d_in[10];
    const float* Wo = (const float*)d_in[11]; const float* bo = (const float*)d_in[12];
    const float* gq = (const float*)d_in[13]; const float* gk = (const float*)d_in[14];

    float* out = (float*)d_out;

    // workspace layout: qh | kh | vh (f16 head-major) | attn (f32 [L,D])
    _Float16* qh = (_Float16*)d_ws;
    _Float16* kh = qh + (size_t)L_ * D_;
    _Float16* vh = kh + (size_t)L_ * D_;
    float* attn  = (float*)(vh + (size_t)L_ * D_);
    float* raw   = out;   // d_out doubles as f32 projection scratch

    dim3 gGrid(D_ / 128, L_ / 128);   // (12, 32)
    dim3 blk(256);

    // Q
    gemm_f16wmma_kernel<<<gGrid, blk, 0, stream>>>(x, Wq, bq, raw, L_, D_, D_);
    rmsrope_kernel<<<L_, blk, 0, stream>>>(raw, gq, fcos, fsin, gs, qh, 1);
    // K
    gemm_f16wmma_kernel<<<gGrid, blk, 0, stream>>>(x, Wk, bk, raw, L_, D_, D_);
    rmsrope_kernel<<<L_, blk, 0, stream>>>(raw, gk, fcos, fsin, gs, kh, 1);
    // V (no norm / no rope, just layout + f16)
    gemm_f16wmma_kernel<<<gGrid, blk, 0, stream>>>(x, Wv, bv, raw, L_, D_, D_);
    rmsrope_kernel<<<L_, blk, 0, stream>>>(raw, gq, fcos, fsin, gs, vh, 0);

    // attention
    dim3 fGrid(L_ / 128, NH);         // (32, 12)
    flash_kernel<<<fGrid, blk, 0, stream>>>(qh, kh, vh, seql, attn);

    // output projection -> d_out
    gemm_f16wmma_kernel<<<gGrid, blk, 0, stream>>>(attn, Wo, bo, out, L_, D_, D_);
}
